// LinearStateLayer_37014028157369
// MI455X (gfx1250) — matrix operand order repaired and checked by reading, not compiled, not run on hardware
//
#include <hip/hip_runtime.h>
#include <hip/hip_bf16.h>
#include <stdint.h>

#define BB 8
#define SS 2048
#define HH 1024
#define NN 256
#define MM (BB * SS)          // 16384 rows
#define LN_EPS 1e-5f

typedef __attribute__((ext_vector_type(16))) __bf16 v16bf;
typedef __attribute__((ext_vector_type(8)))  __bf16 v8bf;
typedef __attribute__((ext_vector_type(8)))  float  v8f;

// ---------------------------------------------------------------------------
// WMMA helpers (gfx1250, wave32, 16x16x32 bf16 -> f32)
// ---------------------------------------------------------------------------
__device__ __forceinline__ v8f wmma_bf16(v16bf a, v16bf b, v8f c) {
  return __builtin_amdgcn_wmma_f32_16x16x32_bf16(
      /*neg_a=*/false, a, /*neg_b=*/false, b,
      /*c_mod=*/(short)0, c, /*reuse_a=*/false, /*reuse_b=*/false);
}

// A operand (16x32 bf16) loaded straight from a row-major [M][K] matrix.
// ISA layout: lane l holds row (l&15); K halves (l>>4)*8 + {0..7} in elems
// 0..7 and (l>>4)*8 + {16..23} in elems 8..15.  Two 16B global loads; all
// waves of a block read the same tile -> L0 resident, no barriers needed.
__device__ __forceinline__ v16bf load_a_frag_g(const __bf16* __restrict__ A,
                                               int stride, int mbase, int ks,
                                               int lane) {
  int row = mbase + (lane & 15);
  int kb  = ks + (lane >> 4) * 8;
  const __bf16* p = A + (size_t)row * stride + kb;
  v8bf lo = *(const v8bf*)(p);
  v8bf hi = *(const v8bf*)(p + 16);
  return __builtin_shufflevector(lo, hi, 0, 1, 2, 3, 4, 5, 6, 7,
                                         8, 9, 10, 11, 12, 13, 14, 15);
}

// B operand: weights stored [out][K] row-major => B-operand column `out` is a
// contiguous run of K. ISA layout: lane l holds column (l&15); K halves
// (l>>4)*8 + {0..7} in elems 0..7 and (l>>4)*8 + {16..23} in elems 8..15.
__device__ __forceinline__ v16bf load_b_frag(const __bf16* __restrict__ W,
                                             int row_stride, int colbase,
                                             int ks, int lane) {
  int col = colbase + (lane & 15);
  int kb  = ks + (lane >> 4) * 8;
  const __bf16* p = W + (size_t)col * row_stride + kb;
  v8bf lo = *(const v8bf*)(p);
  v8bf hi = *(const v8bf*)(p + 16);
  return __builtin_shufflevector(lo, hi, 0, 1, 2, 3, 4, 5, 6, 7,
                                         8, 9, 10, 11, 12, 13, 14, 15);
}

// ---------------------------------------------------------------------------
// f32 -> bf16 conversion (x and all weight matrices)
// ---------------------------------------------------------------------------
__global__ void cvt_f32_bf16(const float* __restrict__ in,
                             __bf16* __restrict__ out, int n) {
  int i = blockIdx.x * blockDim.x + threadIdx.x;
  int stride = gridDim.x * blockDim.x;
  for (; i < n; i += stride) out[i] = (__bf16)in[i];
}

// ---------------------------------------------------------------------------
// Kernel 1: u = sigmoid(x @ gate_w^T + gate_b) * (x @ B_w^T)
// Output written TRANSPOSED as u_t[b][n][s] so the scan can stream it.
// Block = 32 rows x 256 cols; wave (wm = wave&1, wn = wave>>1) owns
// 1 m-tile x 4 n-tiles x {gate,u} = 8 accumulators, 8 WMMA per K-step.
// ---------------------------------------------------------------------------
__global__ __launch_bounds__(256) void proj_gate_kernel(
    const __bf16* __restrict__ xb, const __bf16* __restrict__ gwb,
    const __bf16* __restrict__ bwb, const float* __restrict__ gate_b,
    float* __restrict__ ub_t) {
  int wave = threadIdx.x >> 5, lane = threadIdx.x & 31;
  int wm = wave & 1, wn = wave >> 1;      // wn in 0..3
  int m0 = blockIdx.x * 32;
  int mb = m0 + wm * 16;
  v8f accg[4] = {}, accu[4] = {};

  for (int ks = 0; ks < HH; ks += 32) {
    v16bf a = load_a_frag_g(xb, HH, mb, ks, lane);
#pragma unroll
    for (int j = 0; j < 4; j++) {
      int nb = (wn * 4 + j) * 16;
      v16bf bg = load_b_frag(gwb, HH, nb, ks, lane);
      v16bf bu = load_b_frag(bwb, HH, nb, ks, lane);
      accg[j] = wmma_bf16(a, bg, accg[j]);
      accu[j] = wmma_bf16(a, bu, accu[j]);
    }
  }

  // C/D layout: lane holds col (lane&15), rows (lane>>4)*8 + r -> the 8
  // values of a lane are 8 CONSECUTIVE timesteps => one 32B store per tile.
  int col = lane & 15, rbase = (lane >> 4) * 8;
  int b  = m0 / SS;
  int s0 = (m0 % SS) + wm * 16 + rbase;
#pragma unroll
  for (int j = 0; j < 4; j++) {
    int n = (wn * 4 + j) * 16 + col;
    float gb = gate_b[n];
    v8f res;
#pragma unroll
    for (int r = 0; r < 8; r++) {
      float g = 1.0f / (1.0f + expf(-(accg[j][r] + gb)));  // sigmoid in [0,1]
      res[r] = g * accu[j][r];
    }
    *(v8f*)&ub_t[((size_t)b * NN + n) * SS + s0] = res;
  }
}

// ---------------------------------------------------------------------------
// Kernel 2: clipped scan (non-associative due to clip -> sequential over S).
// One thread per (b,n); u_t[b][n][s] is contiguous per thread -> float4
// loads cover 4 timesteps each.  Emits states[s][n] bf16 (A operand of the
// next GEMM) + final_state (f32 output).
// ---------------------------------------------------------------------------
__global__ __launch_bounds__(256) void scan_kernel(
    const float* __restrict__ u_t, const float* __restrict__ state0,
    const float* __restrict__ A_log, __bf16* __restrict__ states,
    float* __restrict__ final_state) {
  int idx = blockIdx.x * blockDim.x + threadIdx.x;  // 0 .. BB*NN-1
  int b = idx / NN, n = idx % NN;
  float A = expf(A_log[n]);
  A = fminf(fmaxf(A, 0.5f), 0.99f);
  float s = state0[b * NN + n];
  const float* up = u_t + ((size_t)b * NN + n) * SS;
  __bf16* sp = states + (size_t)b * SS * NN + n;
  for (int t = 0; t < SS; t += 4) {
    float4 u4 = *(const float4*)(up + t);
    s = fminf(fmaxf(A * s + u4.x, -10.0f), 10.0f);
    sp[(size_t)(t + 0) * NN] = (__bf16)s;
    s = fminf(fmaxf(A * s + u4.y, -10.0f), 10.0f);
    sp[(size_t)(t + 1) * NN] = (__bf16)s;
    s = fminf(fmaxf(A * s + u4.z, -10.0f), 10.0f);
    sp[(size_t)(t + 2) * NN] = (__bf16)s;
    s = fminf(fmaxf(A * s + u4.w, -10.0f), 10.0f);
    sp[(size_t)(t + 3) * NN] = (__bf16)s;
  }
  final_state[b * NN + n] = s;
}

// ---------------------------------------------------------------------------
// Kernel 3: h = states @ C_w^T + D*x + x ; hn = LayerNorm(h)*g + b  (bf16 out)
// Block owns 16 rows x FULL H=1024 so LN statistics stay in-block:
// 8 waves x 8 h-tiles, K = NN = 256 (8 K-steps).  LDS used only for the
// cross-wave LN reduction.
// ---------------------------------------------------------------------------
__global__ __launch_bounds__(256) void cproj_ln_kernel(
    const __bf16* __restrict__ stb, const __bf16* __restrict__ cwb,
    const float* __restrict__ x, const float* __restrict__ Dv,
    const float* __restrict__ ln_g, const float* __restrict__ ln_b,
    __bf16* __restrict__ hnb) {
  __shared__ float s_sum[16], s_sq[16];
  int wave = threadIdx.x >> 5, lane = threadIdx.x & 31;
  int m0 = blockIdx.x * 16;
  if (threadIdx.x < 16) { s_sum[threadIdx.x] = 0.f; s_sq[threadIdx.x] = 0.f; }
  __syncthreads();

  v8f acc[8] = {};
  for (int ks = 0; ks < NN; ks += 32) {
    v16bf a = load_a_frag_g(stb, NN, m0, ks, lane);
#pragma unroll
    for (int t = 0; t < 8; t++) {
      int hb = (wave * 8 + t) * 16;
      v16bf bm = load_b_frag(cwb, NN, hb, ks, lane);
      acc[t] = wmma_bf16(a, bm, acc[t]);
    }
  }

  int col = lane & 15, rbase = (lane >> 4) * 8;
  float psum[8] = {}, psq[8] = {};
#pragma unroll
  for (int t = 0; t < 8; t++) {
    int hc = (wave * 8 + t) * 16 + col;
    float dv = Dv[hc];
#pragma unroll
    for (int r = 0; r < 8; r++) {
      float xv = x[(size_t)(m0 + rbase + r) * HH + hc];
      float h = acc[t][r] + dv * xv + xv;
      acc[t][r] = h;
      psum[r] += h;
      psq[r] += h * h;
    }
  }
  // lanes 0..15 hold rows 0..7, lanes 16..31 hold rows 8..15 -> reduce in
  // groups of 16, then combine the 8 waves through LDS atomics.
#pragma unroll
  for (int r = 0; r < 8; r++) {
#pragma unroll
    for (int off = 1; off < 16; off <<= 1) {
      psum[r] += __shfl_xor(psum[r], off, 16);
      psq[r]  += __shfl_xor(psq[r],  off, 16);
    }
  }
  if ((lane & 15) == 0) {
#pragma unroll
    for (int r = 0; r < 8; r++) {
      atomicAdd(&s_sum[rbase + r], psum[r]);
      atomicAdd(&s_sq[rbase + r],  psq[r]);
    }
  }
  __syncthreads();

#pragma unroll
  for (int t = 0; t < 8; t++) {
    int hc = (wave * 8 + t) * 16 + col;
    float g = ln_g[hc], bb = ln_b[hc];
#pragma unroll
    for (int r = 0; r < 8; r++) {
      int row = rbase + r;
      float mu  = s_sum[row] * (1.0f / HH);
      float var = s_sq[row] * (1.0f / HH) - mu * mu;
      float rs  = rsqrtf(var + LN_EPS);
      float hn  = (acc[t][r] - mu) * rs * g + bb;
      hnb[(size_t)(m0 + row) * HH + hc] = (__bf16)hn;
    }
  }
}

// ---------------------------------------------------------------------------
// Kernel 4: out = hn @ out_w^T + out_b   [MM x HH], K = HH = 1024
// Grid (MM/32, HH/512); wave (wm = wave&1, wn = wave>>1) owns
// 1 m-tile x 8 o-tiles = 8 WMMA per K-step per A-fragment.
// ---------------------------------------------------------------------------
__global__ __launch_bounds__(256) void out_proj_kernel(
    const __bf16* __restrict__ hnb, const __bf16* __restrict__ owb,
    const float* __restrict__ out_b, float* __restrict__ out) {
  int wave = threadIdx.x >> 5, lane = threadIdx.x & 31;
  int wm = wave & 1, wn = wave >> 1;      // wn in 0..3
  int mb = blockIdx.x * 32 + wm * 16;
  int n0 = blockIdx.y * 512;
  v8f acc[8] = {};

  for (int ks = 0; ks < HH; ks += 32) {
    v16bf a = load_a_frag_g(hnb, HH, mb, ks, lane);
    if (ks + 32 < HH) {  // hint the next K-step's A row into cache
      __builtin_prefetch(
          &hnb[(size_t)(mb + (lane & 15)) * HH + ks + 32 + (lane >> 4) * 8], 0,
          1);
    }
#pragma unroll
    for (int j = 0; j < 8; j++) {
      int ob = n0 + (wn * 8 + j) * 16;
      v16bf bm = load_b_frag(owb, HH, ob, ks, lane);
      acc[j] = wmma_bf16(a, bm, acc[j]);
    }
  }

  int col = lane & 15, rbase = (lane >> 4) * 8;
#pragma unroll
  for (int j = 0; j < 8; j++) {
    int o = n0 + (wn * 8 + j) * 16 + col;
    float ob = out_b[o];
#pragma unroll
    for (int r = 0; r < 8; r++) {
      out[(size_t)(mb + rbase + r) * HH + o] = acc[j][r] + ob;
    }
  }
}

// ---------------------------------------------------------------------------
// Workspace layout (bytes).  hn (bf16, 32MB) overlays xb, which is dead after
// kernel 1.  Total footprint ~60 MB.
// ---------------------------------------------------------------------------
#define OFF_XB 0u                     // x bf16: 33,554,432  (also hn later)
#define OFF_U  33554432u              // u_t f32: 16,777,216
#define OFF_ST 50331648u              // states bf16: 8,388,608
#define OFF_GW 58720256u              // gate_w bf16: 524,288
#define OFF_BW 59244544u              // B_w bf16: 524,288
#define OFF_CW 59768832u              // C_w bf16: 524,288
#define OFF_OW 60293120u              // out_w bf16: 2,097,152

extern "C" void kernel_launch(void* const* d_in, const int* in_sizes, int n_in,
                              void* d_out, int out_size, void* d_ws,
                              size_t ws_size, hipStream_t stream) {
  const float* x      = (const float*)d_in[0];
  const float* state0 = (const float*)d_in[1];
  const float* A_log  = (const float*)d_in[2];
  const float* B_w    = (const float*)d_in[3];
  const float* C_w    = (const float*)d_in[4];
  const float* Dv     = (const float*)d_in[5];
  const float* gate_w = (const float*)d_in[6];
  const float* gate_b = (const float*)d_in[7];
  const float* out_w  = (const float*)d_in[8];
  const float* out_b  = (const float*)d_in[9];
  const float* ln_g   = (const float*)d_in[10];
  const float* ln_b   = (const float*)d_in[11];

  char* ws = (char*)d_ws;
  __bf16* xb  = (__bf16*)(ws + OFF_XB);
  float*  ubt = (float*)(ws + OFF_U);
  __bf16* stb = (__bf16*)(ws + OFF_ST);
  __bf16* gwb = (__bf16*)(ws + OFF_GW);
  __bf16* bwb = (__bf16*)(ws + OFF_BW);
  __bf16* cwb = (__bf16*)(ws + OFF_CW);
  __bf16* owb = (__bf16*)(ws + OFF_OW);
  __bf16* hnb = (__bf16*)(ws + OFF_XB);  // overlay: xb dead after kernel 1

  float* outp = (float*)d_out;
  float* fsp  = outp + (size_t)MM * HH;  // final_state appended after out

  // 0) f32 -> bf16 conversions
  cvt_f32_bf16<<<8192, 256, 0, stream>>>(x, xb, MM * HH);
  cvt_f32_bf16<<<1024, 256, 0, stream>>>(gate_w, gwb, NN * HH);
  cvt_f32_bf16<<<1024, 256, 0, stream>>>(B_w, bwb, NN * HH);
  cvt_f32_bf16<<<1024, 256, 0, stream>>>(C_w, cwb, HH * NN);
  cvt_f32_bf16<<<4096, 256, 0, stream>>>(out_w, owb, HH * HH);

  // 1) gated input projection (WMMA), u written transposed [b][n][s]
  proj_gate_kernel<<<MM / 32, 256, 0, stream>>>(xb, gwb, bwb, gate_b, ubt);

  // 2) sequential clipped scan, one thread per (b, n), float4 u loads
  scan_kernel<<<(BB * NN) / 256, 256, 0, stream>>>(ubt, state0, A_log, stb,
                                                   fsp);

  // 3) C-projection + residual + LayerNorm (WMMA, fused)
  cproj_ln_kernel<<<MM / 16, 256, 0, stream>>>(stb, cwb, x, Dv, ln_g, ln_b,
                                               hnb);

  // 4) output projection (WMMA)
  out_proj_kernel<<<dim3(MM / 32, HH / 512), 256, 0, stream>>>(hnb, owb, out_b,
                                                               outp);
}
